// RSSM_88321707475545
// MI455X (gfx1250) — compile-verified, matrix operand order checked by reading, not run on hardware
//
#include <hip/hip_runtime.h>
#include <hip/hip_bf16.h>

typedef __attribute__((ext_vector_type(16))) _Float16 v16h;
typedef __attribute__((ext_vector_type(8)))  _Float16 v8h;
typedef __attribute__((ext_vector_type(2)))  _Float16 h2;
typedef __attribute__((ext_vector_type(8)))  float    v8f;

// ---------------------------------------------------------------------------
// states f32 (F,3,64,64) -> x0 f16 (F,4,66,66): 1-px zero halo + zero 4th
// channel so conv1 needs no bounds/K guards (K padded 48 -> 64).
// ---------------------------------------------------------------------------
__global__ void prep_input(const float* __restrict__ s, _Float16* __restrict__ x0,
                           long long total) {
    long long idx = (long long)blockIdx.x * blockDim.x + threadIdx.x;
    if (idx >= total) return;
    int pw = (int)(idx % 66);
    long long t = idx / 66;
    int ph = (int)(t % 66); t /= 66;
    int c = (int)(t & 3);
    long long f = t >> 2;
    float v = 0.f;
    if (c < 3 && ph >= 1 && ph <= 64 && pw >= 1 && pw <= 64)
        v = s[((f * 3 + c) * 64 + (ph - 1)) * 64 + (pw - 1)];
    x0[idx] = (_Float16)v;
}

// ---------------------------------------------------------------------------
// Conv weights (Cout, Cin, 4, 4) f32 -> per-lane WMMA B-fragment layout:
// wf[((kt*NTtot + ntile)*32 + lane)*16 + e], K index = ci*16 + kh*4 + kw,
// zero-padded to CinP channels. Lane<16 holds K kofs 0-7/16-23, lane>=16
// holds 8-15/24-31 (mirrors the documented 16-bit A layout).
// ---------------------------------------------------------------------------
__global__ void pack_conv_wfrag(const float* __restrict__ w, _Float16* __restrict__ wf,
                                int Cout, int Cin, int CinP) {
    int idx = blockIdx.x * blockDim.x + threadIdx.x;
    int NTtot = Cout >> 4, Ktiles = CinP >> 1;
    if (idx >= Ktiles * NTtot * 512) return;
    int e = idx & 15, lane = (idx >> 4) & 31, fn = idx >> 9;
    int ntile = fn % NTtot, kt = fn / NTtot;
    int kofs = (lane < 16) ? 0 : 8;
    int k = kt * 32 + ((e & 8) << 1) + kofs + (e & 7);
    int col = ntile * 16 + (lane & 15);
    int ci = k >> 4, r = k & 15;
    float v = (ci < Cin) ? w[(col * Cin + ci) * 16 + r] : 0.f;
    wf[idx] = (_Float16)v;
}

// fc_w (N, K) f32 -> same fragment layout over K tiles of 32
__global__ void pack_fc_wfrag(const float* __restrict__ w, _Float16* __restrict__ wf,
                              int N, int K) {
    int idx = blockIdx.x * blockDim.x + threadIdx.x;
    int NTtot = N >> 4, Ktiles = K >> 5;
    if (idx >= Ktiles * NTtot * 512) return;
    int e = idx & 15, lane = (idx >> 4) & 31, fn = idx >> 9;
    int ntile = fn % NTtot, kt = fn / NTtot;
    int k = kt * 32 + ((e & 8) << 1) + ((lane < 16) ? 0 : 8) + (e & 7);
    int col = ntile * 16 + (lane & 15);
    wf[idx] = (_Float16)w[col * K + k];
}

// ---------------------------------------------------------------------------
// Implicit-GEMM conv (k=4,s=2,p=1) on zero-haloed f16 activations.
// One wave: 16 rows (M) x NT*16 cols (Cout), all guards removed.
// A fragment per k-tile: 4 rows of 4 contiguous halves (2 kh-rows x 2 chans),
// loaded as aligned b32 pairs. B: one aligned v16h per ntile.
// ---------------------------------------------------------------------------
template <int NT>
__global__ void conv_gemm_wmma(const _Float16* __restrict__ in,  // (F,CinP,IHP,IWP)
                               const _Float16* __restrict__ wf,  // fragment packed
                               const float* __restrict__ bias,   // (Cout)
                               _Float16* __restrict__ out,       // (F,Cout,OHP,OWP)
                               int CinP, int IHP, int IWP,
                               int Cout, int lgOW, int lgHW,
                               int OWP, int opad, int groups, long long Mtiles) {
    const int lane = threadIdx.x & 31;
    const long long tile = (long long)blockIdx.x * (blockDim.x >> 5) + (threadIdx.x >> 5);
    if (tile >= Mtiles * groups) return;           // wave-uniform exit
    const int ng = (int)(tile % groups);
    const long long mt = tile / groups;

    const int OW = 1 << lgOW;
    const int hwMask = (1 << lgHW) - 1;
    const int NTtot = Cout >> 4;

    const long long m = mt * 16 + (lane & 15);
    const int sp = (int)m & hwMask;
    const long long frame = m >> lgHW;
    const int oh = sp >> lgOW, ow = sp & (OW - 1);
    const int khb = (lane < 16) ? 0 : 2;           // kh rows this half-wave owns
    const int chanStride = IHP * IWP;
    const _Float16* fbase = in + frame * (long long)CinP * chanStride
                               + (long long)(oh * 2 + khb) * IWP + ow * 2;

    v8f zero = {};
    v8f acc[NT];
#pragma unroll
    for (int nt = 0; nt < NT; ++nt) acc[nt] = zero;

    const int Ktiles = CinP >> 1;
    for (int kt = 0; kt < Ktiles; ++kt) {
        const _Float16* p0 = fbase + (long long)(2 * kt) * chanStride;
        v16h a;
#pragma unroll
        for (int j = 0; j < 4; ++j) {              // (ci+0/1) x (khb, khb+1)
            const _Float16* pr = p0 + ((j & 1) ? IWP : 0) + ((j & 2) ? chanStride : 0);
            const unsigned int* pu = (const unsigned int*)pr;  // 4B-aligned (halo parity)
            h2 lo = __builtin_bit_cast(h2, pu[0]);
            h2 hi = __builtin_bit_cast(h2, pu[1]);
            a[4 * j + 0] = lo[0]; a[4 * j + 1] = lo[1];
            a[4 * j + 2] = hi[0]; a[4 * j + 3] = hi[1];
        }
        const _Float16* wp = wf + ((long long)(kt * NTtot + ng * NT) * 32 + lane) * 16;
#pragma unroll
        for (int nt = 0; nt < NT; ++nt) {
            v16h b = *(const v16h*)(wp + nt * 512);
            acc[nt] = __builtin_amdgcn_wmma_f32_16x16x32_f16(
                          false, a, false, b, (short)0, acc[nt], false, false);
        }
    }

    const long long mr0 = mt * 16 + ((lane >> 4) << 3);   // C: VGPR v -> row v(+8)
    const long long OHP = (long long)(1 << (lgHW - lgOW)) + 2 * opad;
#pragma unroll
    for (int nt = 0; nt < NT; ++nt) {
        const int col = (ng * NT + nt) * 16 + (lane & 15);
        const float bb = bias[col];
#pragma unroll
        for (int v = 0; v < 8; ++v) {
            long long mm = mr0 + v;
            int sp2 = (int)mm & hwMask;
            long long f2 = mm >> lgHW;
            int oh2 = sp2 >> lgOW, ow2 = sp2 & (OW - 1);
            float val = fmaxf(acc[nt][v] + bb, 0.f);
            out[((f2 * Cout + col) * OHP + oh2 + opad) * OWP + ow2 + opad] = (_Float16)val;
        }
    }
}

// ---------------------------------------------------------------------------
// FC head: (M,K) f16 x packed (K,N) f16 + bias -> f32, NT ntiles per wave.
// ---------------------------------------------------------------------------
template <int NT>
__global__ void fc_gemm_wmma(const _Float16* __restrict__ A,
                             const _Float16* __restrict__ wf,
                             const float* __restrict__ bias,
                             float* __restrict__ out, int M, int N, int K) {
    const int lane = threadIdx.x & 31;
    const int NTtot = N >> 4;
    const int groups = NTtot / NT;
    const int tile = blockIdx.x * (blockDim.x >> 5) + (threadIdx.x >> 5);
    if (tile >= (M >> 4) * groups) return;
    const int ng = tile % groups;
    const int mt = tile / groups;
    const int row = mt * 16 + (lane & 15);
    const int kofs = (lane < 16) ? 0 : 8;
    const _Float16* arow = A + (long long)row * K + kofs;

    v8f zero = {};
    v8f acc[NT];
#pragma unroll
    for (int nt = 0; nt < NT; ++nt) acc[nt] = zero;

    for (int k0 = 0; k0 < K; k0 += 32) {
        v8h alo = *(const v8h*)(arow + k0);        // 16B-aligned
        v8h ahi = *(const v8h*)(arow + k0 + 16);
        v16h a;
#pragma unroll
        for (int e = 0; e < 8; ++e) { a[e] = alo[e]; a[8 + e] = ahi[e]; }
        const _Float16* wp = wf + ((long long)((k0 >> 5) * NTtot + ng * NT) * 32 + lane) * 16;
#pragma unroll
        for (int nt = 0; nt < NT; ++nt) {
            v16h b = *(const v16h*)(wp + nt * 512);
            acc[nt] = __builtin_amdgcn_wmma_f32_16x16x32_f16(
                          false, a, false, b, (short)0, acc[nt], false, false);
        }
    }
    const int r0 = mt * 16 + ((lane >> 4) << 3);
#pragma unroll
    for (int nt = 0; nt < NT; ++nt) {
        const int col = (ng * NT + nt) * 16 + (lane & 15);
        const float bb = bias[col];
#pragma unroll
        for (int v = 0; v < 8; ++v)
            out[(long long)(r0 + v) * N + col] = acc[nt][v] + bb;
    }
}

// ---------------------------------------------------------------------------
// Sequential RSSM scan: one workgroup per batch row, state in LDS, f32 VALU.
// ---------------------------------------------------------------------------
__global__ void rssm_recurrent(const float* __restrict__ actions, // (B,T,A)
                               const float* __restrict__ feats,   // (B*T, FD)
                               const float* __restrict__ wih,     // (3H, L+A)
                               const float* __restrict__ whh,     // (3H, H)
                               const float* __restrict__ bih,
                               const float* __restrict__ bhh,
                               const float* __restrict__ prw,     // (2L, H)
                               const float* __restrict__ prb,
                               const float* __restrict__ pow_,    // (2L, H+FD)
                               const float* __restrict__ pob,
                               const float* __restrict__ eps,     // (B,T,L)
                               float* __restrict__ outp) {
    const int B = 32, T = 64, A = 6, L = 32, H = 128, FD = 256;
    const int XD = L + A;
    const int b = blockIdx.x;
    const int tid = threadIdx.x;

    __shared__ float sh[128], shn[128], sz[32], sx[38], sgi[384], sgh[384];
    if (tid < H) sh[tid] = 0.f;
    if (tid < L) sz[tid] = 0.f;
    __syncthreads();

    const long long O_lvp = (long long)B * T * L;
    const long long O_muq = 2 * O_lvp, O_lvq = 3 * O_lvp;
    const long long O_h   = 4 * O_lvp, O_z = O_h + (long long)B * T * H;

    for (int t = 0; t < T; ++t) {
        const int bt = b * T + t;
        if (tid < XD) sx[tid] = (tid < L) ? sz[tid] : actions[bt * A + (tid - L)];
        __syncthreads();

        for (int j = tid; j < 3 * H; j += blockDim.x) {
            float s = bih[j];
            for (int k = 0; k < XD; ++k) s += sx[k] * wih[j * XD + k];
            sgi[j] = s;
            float s2 = bhh[j];
            for (int k = 0; k < H; ++k) s2 += sh[k] * whh[j * H + k];
            sgh[j] = s2;
        }
        __syncthreads();

        if (tid < H) {
            float r  = 1.f / (1.f + __expf(-(sgi[tid] + sgh[tid])));
            float zg = 1.f / (1.f + __expf(-(sgi[H + tid] + sgh[H + tid])));
            float n  = tanhf(sgi[2 * H + tid] + r * sgh[2 * H + tid]);
            shn[tid] = (1.f - zg) * n + zg * sh[tid];
        }
        __syncthreads();
        if (tid < H) sh[tid] = shn[tid];
        __syncthreads();

        const float* f = feats + (long long)bt * FD;
        if (tid < 2 * L) {
            float s = prb[tid];
            for (int k = 0; k < H; ++k) s += prw[tid * H + k] * sh[k];
            long long o = (long long)bt * L + (tid & (L - 1));
            if (tid < L) outp[o] = s; else outp[O_lvp + o] = s;
        } else {
            int i = tid - 2 * L;
            float s = pob[i];
            for (int k = 0; k < H;  ++k) s += pow_[i * (H + FD) + k] * sh[k];
            for (int k = 0; k < FD; ++k) s += pow_[i * (H + FD) + H + k] * f[k];
            sgi[i] = s;
            long long o = (long long)bt * L + (i & (L - 1));
            if (i < L) outp[O_muq + o] = s; else outp[O_lvq + o] = s;
        }
        __syncthreads();

        if (tid < L) {
            float zv = sgi[tid] + __expf(0.5f * sgi[L + tid]) * eps[bt * L + tid];
            sz[tid] = zv;
            outp[O_z + (long long)bt * L + tid] = zv;
        }
        if (tid < H) outp[O_h + (long long)bt * H + tid] = sh[tid];
        __syncthreads();
    }
}

// ---------------------------------------------------------------------------
extern "C" void kernel_launch(void* const* d_in, const int* in_sizes, int n_in,
                              void* d_out, int out_size, void* d_ws, size_t ws_size,
                              hipStream_t stream) {
    (void)in_sizes; (void)n_in; (void)out_size; (void)ws_size;
    const float* states  = (const float*)d_in[0];
    const float* actions = (const float*)d_in[1];
    const float* c1w = (const float*)d_in[2];  const float* c1b = (const float*)d_in[3];
    const float* c2w = (const float*)d_in[4];  const float* c2b = (const float*)d_in[5];
    const float* c3w = (const float*)d_in[6];  const float* c3b = (const float*)d_in[7];
    const float* c4w = (const float*)d_in[8];  const float* c4b = (const float*)d_in[9];
    const float* fcw = (const float*)d_in[10]; const float* fcb = (const float*)d_in[11];
    const float* wih = (const float*)d_in[12]; const float* whh = (const float*)d_in[13];
    const float* bih = (const float*)d_in[14]; const float* bhh = (const float*)d_in[15];
    const float* prw = (const float*)d_in[16]; const float* prb = (const float*)d_in[17];
    const float* pow_ = (const float*)d_in[18]; const float* pob = (const float*)d_in[19];
    const float* eps = (const float*)d_in[20];
    float* outp = (float*)d_out;

    const long long F = 32LL * 64;                       // 2048 frames
    char*  ws  = (char*)d_ws;
    size_t off = 0;
    auto alloc = [&](size_t bytes) {
        void* p = ws + off;
        off += (bytes + 255) & ~(size_t)255;
        return p;
    };
    const size_t x0_n = (size_t)F * 4 * 66 * 66;
    const size_t a1_n = (size_t)F * 32 * 34 * 34;
    const size_t a2_n = (size_t)F * 64 * 18 * 18;
    const size_t a3_n = (size_t)F * 128 * 10 * 10;
    const size_t a4_n = (size_t)F * 256 * 16;
    _Float16* x0  = (_Float16*)alloc(x0_n * 2);
    _Float16* a1  = (_Float16*)alloc(a1_n * 2);
    _Float16* a2  = (_Float16*)alloc(a2_n * 2);
    _Float16* a3  = (_Float16*)alloc(a3_n * 2);
    _Float16* a4  = (_Float16*)alloc(a4_n * 2);
    float*    ft  = (float*)   alloc((size_t)F * 256 * 4);
    _Float16* w1f = (_Float16*)alloc((size_t)2  * 2  * 512 * 2);   // Ktiles*NTtot*512
    _Float16* w2f = (_Float16*)alloc((size_t)16 * 4  * 512 * 2);
    _Float16* w3f = (_Float16*)alloc((size_t)32 * 8  * 512 * 2);
    _Float16* w4f = (_Float16*)alloc((size_t)64 * 16 * 512 * 2);
    _Float16* wff = (_Float16*)alloc((size_t)128 * 16 * 512 * 2);

    auto cdiv = [](long long a, long long b) { return (int)((a + b - 1) / b); };

    // zero halos of padded activation buffers (conv writes interiors only)
    hipMemsetAsync(a1, 0, a1_n * 2, stream);
    hipMemsetAsync(a2, 0, a2_n * 2, stream);
    hipMemsetAsync(a3, 0, a3_n * 2, stream);

    // input prep + weight packing
    prep_input<<<cdiv((long long)x0_n, 256), 256, 0, stream>>>(states, x0, (long long)x0_n);
    pack_conv_wfrag<<<cdiv(2 * 2 * 512, 256),   256, 0, stream>>>(c1w, w1f, 32, 3, 4);
    pack_conv_wfrag<<<cdiv(16 * 4 * 512, 256),  256, 0, stream>>>(c2w, w2f, 64, 32, 32);
    pack_conv_wfrag<<<cdiv(32 * 8 * 512, 256),  256, 0, stream>>>(c3w, w3f, 128, 64, 64);
    pack_conv_wfrag<<<cdiv(64 * 16 * 512, 256), 256, 0, stream>>>(c4w, w4f, 256, 128, 128);
    pack_fc_wfrag<<<cdiv(128 * 16 * 512, 256),  256, 0, stream>>>(fcw, wff, 256, 4096);

    // encoder: implicit-GEMM WMMA convs (8 waves/block; one 16x(NT*16) tile/wave)
    //           CinP IHP IWP Cout lgOW lgHW OWP opad groups Mtiles
    conv_gemm_wmma<2><<<cdiv(131072, 8), 256, 0, stream>>>(
        x0, w1f, c1b, a1, 4, 66, 66, 32, 5, 10, 34, 1, 1, 131072);
    conv_gemm_wmma<4><<<cdiv(32768, 8), 256, 0, stream>>>(
        a1, w2f, c2b, a2, 32, 34, 34, 64, 4, 8, 18, 1, 1, 32768);
    conv_gemm_wmma<8><<<cdiv(8192, 8), 256, 0, stream>>>(
        a2, w3f, c3b, a3, 64, 18, 18, 128, 3, 6, 10, 1, 1, 8192);
    conv_gemm_wmma<8><<<cdiv(2048 * 2, 8), 256, 0, stream>>>(
        a3, w4f, c4b, a4, 128, 10, 10, 256, 2, 4, 4, 0, 2, 2048);

    // FC head -> f32 features
    fc_gemm_wmma<4><<<cdiv(128 * 4, 8), 256, 0, stream>>>(a4, wff, fcb, ft,
                                                          (int)F, 256, 4096);

    // sequential scan: one workgroup per batch row
    rssm_recurrent<<<32, 128, 0, stream>>>(actions, ft, wih, whh, bih, bhh,
                                           prw, prb, pow_, pob, eps, outp);
}
